// CovNet_3058016715001
// MI455X (gfx1250) — matrix operand order repaired and checked by reading, not compiled, not run on hardware
//
#include <hip/hip_runtime.h>
#include <hip/hip_bf16.h>

// ---------------------------------------------------------------------------
// BNN MLP on MI455X (gfx1250, wave32, WMMA):
//   h   = clip(x @ sign(W1)^T + b1, -1, 1)     [8192 x 4096]
//   out = h @ W2^T + b2                        [8192 x 10]
// fc1 (52.6 GFLOP) -> v_wmma_f32_16x16x32_f16 with K padded 784->800 so the
// inner loop is branch-free; fc2 (0.67 GFLOP) -> VALU + wave reduction.
// ---------------------------------------------------------------------------

typedef __attribute__((ext_vector_type(16))) _Float16 v16h;
typedef __attribute__((ext_vector_type(8)))  _Float16 v8h;
typedef __attribute__((ext_vector_type(4)))  _Float16 v4h;
typedef __attribute__((ext_vector_type(8)))  float    v8f;

#define BATCH 8192
#define IN    784
#define KPAD  800          // 25 * 32, zero-padded K for branch-free WMMA loop
#define HID   4096
#define NOUT  10

// fc1 tiling: 8 waves/block; each wave owns a 64x64 output tile (4x4 WMMAs).
#define WAVE_M  64
#define WAVE_N  64
#define BW_M    2
#define BW_N    4
#define BLOCK_M (WAVE_M * BW_M)   // 128
#define BLOCK_N (WAVE_N * BW_N)   // 256

union Frag16 { v16h v; v8h h[2]; };

struct Bases { const _Float16* a[4]; const _Float16* b[4]; };

// --------------------- W1 -> sign() in f16, K-padded -----------------------
__global__ __launch_bounds__(256) void k_binarize_w1(const float* __restrict__ W1,
                                                     _Float16* __restrict__ w1h) {
  int i = (blockIdx.x * 256 + threadIdx.x) * 4;
  if (i >= HID * KPAD) return;
  int row = i / KPAD, k = i % KPAD;      // 4-chunks never straddle the pad
  v4h r = {};
  if (k < IN) {
    float4 w = *(const float4*)(W1 + (size_t)row * IN + k);
    r[0] = (_Float16)((w.x > 0.f) ? 1.f : ((w.x < 0.f) ? -1.f : 0.f));
    r[1] = (_Float16)((w.y > 0.f) ? 1.f : ((w.y < 0.f) ? -1.f : 0.f));
    r[2] = (_Float16)((w.z > 0.f) ? 1.f : ((w.z < 0.f) ? -1.f : 0.f));
    r[3] = (_Float16)((w.w > 0.f) ? 1.f : ((w.w < 0.f) ? -1.f : 0.f));
  }
  *(v4h*)(w1h + i) = r;
}

// -------------------------- x -> f16, K-padded -----------------------------
__global__ __launch_bounds__(256) void k_convert_x(const float* __restrict__ x,
                                                   _Float16* __restrict__ xh) {
  int i = (blockIdx.x * 256 + threadIdx.x) * 4;
  if (i >= BATCH * KPAD) return;
  int row = i / KPAD, k = i % KPAD;
  v4h r = {};
  if (k < IN) {
    float4 w = *(const float4*)(x + (size_t)row * IN + k);
    r[0] = (_Float16)w.x;
    r[1] = (_Float16)w.y;
    r[2] = (_Float16)w.z;
    r[3] = (_Float16)w.w;
  }
  *(v4h*)(xh + i) = r;
}

// ---------------- fragment loads (ISA 7.12.2 operand layouts) --------------
// A (16x32): lanes 0-15 elems 0..7 = K+0..7, 8..15 = K+16..23; lanes 16-31
//            elems 0..7 = K+8..15, 8..15 = K+24..31.  (lhalf*8 folded in base)
// B (32x16): same K layout as A (lanes 0-15: K+0..7,16..23; lanes 16-31:
//            K+8..15,24..31).  Column n of B == row n of w1h. (lhalf*8 in base)
__device__ __forceinline__ void load_frags(const Bases& bs, int kb,
                                           v16h* __restrict__ a,
                                           v16h* __restrict__ b) {
#pragma unroll
  for (int mi = 0; mi < 4; ++mi) {
    Frag16 f;
    f.h[0] = *(const v8h*)(bs.a[mi] + kb);
    f.h[1] = *(const v8h*)(bs.a[mi] + kb + 16);
    a[mi] = f.v;
  }
#pragma unroll
  for (int ni = 0; ni < 4; ++ni) {
    Frag16 f;
    f.h[0] = *(const v8h*)(bs.b[ni] + kb);
    f.h[1] = *(const v8h*)(bs.b[ni] + kb + 16);
    b[ni] = f.v;
  }
}

__device__ __forceinline__ void do_mma(const v16h* __restrict__ a,
                                       const v16h* __restrict__ b,
                                       v8f acc[4][4]) {
#pragma unroll
  for (int mi = 0; mi < 4; ++mi)
#pragma unroll
    for (int ni = 0; ni < 4; ++ni)
      acc[mi][ni] = __builtin_amdgcn_wmma_f32_16x16x32_f16(
          false, a[mi], false, b[ni], (short)0, acc[mi][ni], false, false);
}

// ------------------ fc1: WMMA GEMM + bias + hardtanh -> H ------------------
__global__ __launch_bounds__(256) void k_fc1_wmma(const _Float16* __restrict__ xh,
                                                  const _Float16* __restrict__ w1h,
                                                  const float* __restrict__ b1,
                                                  _Float16* __restrict__ H) {
  const int lane  = threadIdx.x & 31;
  const int lrow  = lane & 15;   // M (A) / N (B,C,D) index
  const int lhalf = lane >> 4;
  const int wave  = threadIdx.x >> 5;

  const int m_base = blockIdx.y * BLOCK_M + (wave / BW_N) * WAVE_M;
  const int n_base = blockIdx.x * BLOCK_N + (wave % BW_N) * WAVE_N;

  Bases bs;
#pragma unroll
  for (int mi = 0; mi < 4; ++mi)
    bs.a[mi] = xh + (size_t)(m_base + mi * 16 + lrow) * KPAD + lhalf * 8;
#pragma unroll
  for (int ni = 0; ni < 4; ++ni)
    bs.b[ni] = w1h + (size_t)(n_base + ni * 16 + lrow) * KPAD + lhalf * 8;

  v8f acc[4][4];
  const v8f zacc = {};
#pragma unroll
  for (int mi = 0; mi < 4; ++mi)
#pragma unroll
    for (int ni = 0; ni < 4; ++ni) acc[mi][ni] = zacc;

  // 25 k-steps, 2-deep software pipeline: preload step 0, then 12 double
  // iterations (load s+1 / mma s), final mma on the last preloaded step.
  v16h a0[4], b0[4], a1[4], b1v[4];
  load_frags(bs, 0, a0, b0);

#pragma unroll 1
  for (int i = 0; i < (KPAD / 32 - 1) / 2; ++i) {   // 12 iterations
    const int kb = 32 + 64 * i;
    __builtin_prefetch(bs.a[0] + kb + 224, 0, 3);   // ~7 k-steps ahead
    __builtin_prefetch(bs.b[0] + kb + 224, 0, 3);
    load_frags(bs, kb, a1, b1v);
    do_mma(a0, b0, acc);
    load_frags(bs, kb + 32, a0, b0);
    do_mma(a1, b1v, acc);
  }
  do_mma(a0, b0, acc);                              // step 24 (k = 768..799)

  // Epilogue: + b1, hardtanh, store f16 H tile.
  // D layout: VGPR v -> (M = v + 8*lhalf, N = lrow) within each 16x16 tile.
#pragma unroll
  for (int ni = 0; ni < 4; ++ni) {
    const int col = n_base + ni * 16 + lrow;
    const float bias = b1[col];
#pragma unroll
    for (int mi = 0; mi < 4; ++mi) {
      const int row0 = m_base + mi * 16 + lhalf * 8;
      v8f a = acc[mi][ni];
#pragma unroll
      for (int v = 0; v < 8; ++v) {
        float t = a[v] + bias;
        t = fminf(1.0f, fmaxf(-1.0f, t));
        H[(size_t)(row0 + v) * HID + col] = (_Float16)t;
      }
    }
  }
}

// --------------------- fc2: out = H @ W2^T + b2 (VALU) ---------------------
__global__ __launch_bounds__(256) void k_fc2(const _Float16* __restrict__ H,
                                             const float* __restrict__ W2,
                                             const float* __restrict__ b2,
                                             float* __restrict__ out) {
  const int lane = threadIdx.x & 31;
  const int wave = threadIdx.x >> 5;
  const int b = blockIdx.x * 8 + wave;   // one wave per batch row

  float acc[NOUT];
#pragma unroll
  for (int o = 0; o < NOUT; ++o) acc[o] = 0.0f;

  const _Float16* hrow = H + (size_t)b * HID;
#pragma unroll 4
  for (int it = 0; it < HID / 256; ++it) {      // 16 iterations
    const int hid0 = it * 256 + lane * 8;       // 8 contiguous elems per lane
    v8h hv = *(const v8h*)(hrow + hid0);
    float hf[8];
#pragma unroll
    for (int i = 0; i < 8; ++i) hf[i] = (float)hv[i];
#pragma unroll
    for (int o = 0; o < NOUT; ++o) {
      const float4* w = (const float4*)(W2 + (size_t)o * HID + hid0);
      float4 w0 = w[0], w1 = w[1];
      acc[o] += hf[0] * w0.x + hf[1] * w0.y + hf[2] * w0.z + hf[3] * w0.w +
                hf[4] * w1.x + hf[5] * w1.y + hf[6] * w1.z + hf[7] * w1.w;
    }
  }

  // Deterministic intra-wave (32-lane) tree reduction.
#pragma unroll
  for (int o = 0; o < NOUT; ++o)
#pragma unroll
    for (int m = 16; m >= 1; m >>= 1) acc[o] += __shfl_xor(acc[o], m, 32);

  if (lane == 0) {
#pragma unroll
    for (int o = 0; o < NOUT; ++o) out[(size_t)b * NOUT + o] = acc[o] + b2[o];
  }
}

// ------------------------------- launcher ----------------------------------
extern "C" void kernel_launch(void* const* d_in, const int* in_sizes, int n_in,
                              void* d_out, int out_size, void* d_ws, size_t ws_size,
                              hipStream_t stream) {
  const float* x  = (const float*)d_in[0];
  const float* W1 = (const float*)d_in[1];
  const float* b1 = (const float*)d_in[2];
  const float* W2 = (const float*)d_in[3];
  const float* b2 = (const float*)d_in[4];
  float* out = (float*)d_out;

  // Workspace layout (all 16B-aligned):
  //   xh  : BATCH*KPAD f16 = 13,107,200 B
  //   w1h : HID*KPAD  f16 =  6,553,600 B
  //   H   : BATCH*HID f16 = 67,108,864 B   (total ~86.8 MB)
  char* ws = (char*)d_ws;
  _Float16* xh  = (_Float16*)ws;
  _Float16* w1h = (_Float16*)(ws + (size_t)BATCH * KPAD * 2);
  _Float16* H   = (_Float16*)(ws + (size_t)BATCH * KPAD * 2 + (size_t)HID * KPAD * 2);

  k_binarize_w1<<<(HID * KPAD / 4 + 255) / 256, 256, 0, stream>>>(W1, w1h);
  k_convert_x<<<(BATCH * KPAD / 4 + 255) / 256, 256, 0, stream>>>(x, xh);

  dim3 grid(HID / BLOCK_N, BATCH / BLOCK_M);  // (16, 64)
  k_fc1_wmma<<<grid, 256, 0, stream>>>(xh, w1h, b1, H);

  k_fc2<<<BATCH / 8, 256, 0, stream>>>(H, W2, b2, out);
}
